// ContextEncoder_65240553226394
// MI455X (gfx1250) — compile-verified, matrix operand order checked by reading, not run on hardware
//
#include <hip/hip_runtime.h>
#include <hip/hip_bf16.h>

// ---------------------------------------------------------------------------
// ChebConv (K=3) GNN, 3 layers: 128 -> 250 -> 500 -> 1000 on gfx1250 (MI455X)
//   prop(h): scatter-add SpMM over 3.2M edges (L2-resident gathers, fp32 atomics)
//   cheb   : t0=h, t1=prop(h), t2=2*prop(t1)-t0 ; out = relu(sum tk@Wk + b)
// Dense GEMM: bf16-packed operands, double-buffered LDS fed by
// global_load_async_to_lds_b128 (ASYNCcnt), v_wmma_f32_16x16x32_bf16 compute.
// ---------------------------------------------------------------------------

typedef __attribute__((ext_vector_type(16))) __bf16        v16bf;
typedef __attribute__((ext_vector_type(8)))  float         v8f;
typedef __attribute__((ext_vector_type(4)))  unsigned int  v4u;

union FragBF16 {            // 16 bf16 per lane = 8 VGPRs
    v16bf v;
    v4u   q[2];
    unsigned int u[8];
};

__device__ __forceinline__ unsigned short f32_to_bf16_rne(float f) {
    unsigned int u = __float_as_uint(f);
    unsigned int r = 0x7FFFu + ((u >> 16) & 1u);   // round-to-nearest-even
    return (unsigned short)((u + r) >> 16);
}

__device__ __forceinline__ void atomic_add_f32(float* p, float v) {
    __hip_atomic_fetch_add(p, v, __ATOMIC_RELAXED, __HIP_MEMORY_SCOPE_AGENT);
}

// Async global->LDS copy, 16B per lane, ASYNCcnt-tracked (CDNA5).
__device__ __forceinline__ void async_copy_b128(unsigned lds_off, unsigned long long gbase,
                                                unsigned goff) {
    asm volatile("global_load_async_to_lds_b128 %0, %1, %2 offset:0"
                 :: "v"(lds_off), "v"(goff), "s"(gbase)
                 : "memory");
}

// ---------------------------------------------------------------------------
// Utility / graph-normalization kernels
// ---------------------------------------------------------------------------

__global__ void fill_zero_kernel(float* __restrict__ p, size_t n) {
    size_t i = (size_t)blockIdx.x * blockDim.x + threadIdx.x;
    size_t stride = (size_t)gridDim.x * blockDim.x;
    for (; i < n; i += stride) p[i] = 0.0f;
}

__global__ void degree_kernel(const int* __restrict__ src, float* __restrict__ deg, int E) {
    int e = blockIdx.x * blockDim.x + threadIdx.x;
    if (e < E) atomic_add_f32(&deg[src[e]], 1.0f);
}

__global__ void dis_kernel(const float* __restrict__ deg, float* __restrict__ dis, int N) {
    int i = blockIdx.x * blockDim.x + threadIdx.x;
    if (i < N) {
        float d = deg[i];
        dis[i] = (d > 0.0f) ? rsqrtf(fmaxf(d, 1.0f)) : 0.0f;
    }
}

__global__ void edge_weight_kernel(const int* __restrict__ src, const int* __restrict__ dst,
                                   const float* __restrict__ dis, float* __restrict__ w, int E) {
    int e = blockIdx.x * blockDim.x + threadIdx.x;
    if (e < E) w[e] = -dis[src[e]] * dis[dst[e]];
}

// ---------------------------------------------------------------------------
// SpMM scatter: out[dst[e], :] += w[e] * in[src[e], :]
// One block per edge; coalesced feature rows; hardware fp32 atomics.
// ---------------------------------------------------------------------------

__global__ __launch_bounds__(256)
void spmm_scatter_kernel(const float* __restrict__ h, float* __restrict__ out,
                         const int* __restrict__ src, const int* __restrict__ dst,
                         const float* __restrict__ w, int F) {
    int e = blockIdx.x;
    int s = src[e];
    int d = dst[e];
    float ww = w[e];
    const float* hr = h + (size_t)s * F;
    float* orow = out + (size_t)d * F;
    for (int f = threadIdx.x; f < F; f += blockDim.x)
        atomic_add_f32(&orow[f], ww * hr[f]);
}

// ---------------------------------------------------------------------------
// Operand packing to bf16 (zero-padded so GEMM hot loop needs no bounds checks)
//   Abf[r][t*Kp + k] = bf16( t==0 ? t0 : t==1 ? t1 : 2*p2 - t0 )   (cheb fused)
//   Bbf[(t*FoutPad + n)*Kp + k] = bf16( W[t][k][n] )               (n-major)
// ---------------------------------------------------------------------------

__global__ void pack_A_kernel(const float* __restrict__ t0, const float* __restrict__ t1,
                              const float* __restrict__ p2, unsigned short* __restrict__ Abf,
                              int N, int Mpad, int K, int Kp) {
    size_t total = (size_t)Mpad * 3 * Kp;
    size_t stride = (size_t)gridDim.x * blockDim.x;
    for (size_t idx = (size_t)blockIdx.x * blockDim.x + threadIdx.x; idx < total; idx += stride) {
        int k = (int)(idx % Kp);
        size_t rest = idx / Kp;
        int t = (int)(rest % 3);
        int r = (int)(rest / 3);
        float v = 0.0f;
        if (r < N && k < K) {
            size_t o = (size_t)r * K + k;
            v = (t == 0) ? t0[o] : (t == 1) ? t1[o] : (2.0f * p2[o] - t0[o]);
        }
        Abf[((size_t)r * 3 + t) * Kp + k] = f32_to_bf16_rne(v);
    }
}

__global__ void pack_B_kernel(const float* __restrict__ W, unsigned short* __restrict__ Bbf,
                              int K, int Kp, int Fout, int FoutPad) {
    size_t total = (size_t)3 * FoutPad * Kp;
    size_t stride = (size_t)gridDim.x * blockDim.x;
    for (size_t idx = (size_t)blockIdx.x * blockDim.x + threadIdx.x; idx < total; idx += stride) {
        int k = (int)(idx % Kp);
        size_t rest = idx / Kp;
        int n = (int)(rest % FoutPad);
        int t = (int)(rest / FoutPad);
        float v = 0.0f;
        if (k < K && n < Fout) v = W[((size_t)t * K + k) * Fout + n];
        Bbf[idx] = f32_to_bf16_rne(v);
    }
}

// ---------------------------------------------------------------------------
// bf16 WMMA GEMM + bias + ReLU:
//   out[N x Fout] = relu( Abf[Mpad x KT] @ Bbf^T + bias )
// Block tile 128x128, 8 waves (wave32), each wave owns 32x64 (2x4 WMMA accs).
// A/B tiles double-buffered in LDS, filled by global_load_async_to_lds_b128.
// ---------------------------------------------------------------------------

#define BM 128
#define BN 128
#define BK 32

__global__ __launch_bounds__(256, 1)
void wmma_gemm_bf16_kernel(const unsigned short* __restrict__ Abf,
                           const unsigned short* __restrict__ Bbf,
                           const float* __restrict__ bias, float* __restrict__ out,
                           int N, int KT, int Kp, int Fout, int FoutPad) {
    __shared__ __align__(16) unsigned short sA[2][BM * BK];   // [buf][m][k]  8KB each
    __shared__ __align__(16) unsigned short sB[2][BN * BK];   // [buf][n][k]  8KB each

    const int tid  = threadIdx.x;
    const int lane = tid & 31;
    const int wid  = tid >> 5;
    const int wm   = wid & 3;      // 4 wave-rows  -> 32 M each
    const int wn   = wid >> 2;     // 2 wave-cols  -> 64 N each
    const int m0   = blockIdx.y * BM;
    const int n0   = blockIdx.x * BN;

    const int mrow  = lane & 15;
    const int khalf = lane >> 4;

    const unsigned long long aBase = (unsigned long long)Abf;
    const unsigned long long bBase = (unsigned long long)Bbf;
    const unsigned ldsA = (unsigned)(unsigned long long)(void*)&sA[0][0];
    const unsigned ldsB = (unsigned)(unsigned long long)(void*)&sB[0][0];

    const int tilesPerTerm = Kp / BK;
    const int totalChunks  = 3 * tilesPerTerm;

    // Issue one 128x32 A tile + 128x32 B tile (8KB each) into LDS buffer `b`.
    // 512 x 16B per tile -> 2 b128 async transfers per thread per tile.
    auto issue = [&](int c, int b) {
        int term = c / tilesPerTerm;
        int kloc = (c - term * tilesPerTerm) * BK;
        #pragma unroll
        for (int i = 0; i < 2; ++i) {
            unsigned s = (unsigned)tid + (unsigned)i * 256u;   // 0..511
            unsigned r = s >> 2;                               // row 0..127
            unsigned q = s & 3;                                // 16B chunk in 64B row
            unsigned aoff = (((unsigned)(m0 + (int)r) * (unsigned)KT + (unsigned)c * BK) << 1)
                            + q * 16u;
            async_copy_b128(ldsA + (unsigned)b * (BM * BK * 2) + s * 16u, aBase, aoff);
            unsigned boff = ((((unsigned)term * (unsigned)FoutPad + (unsigned)(n0 + (int)r))
                              * (unsigned)Kp + (unsigned)kloc) << 1) + q * 16u;
            async_copy_b128(ldsB + (unsigned)b * (BN * BK * 2) + s * 16u, bBase, boff);
        }
    };

    v8f acc[2][4];
    #pragma unroll
    for (int mi = 0; mi < 2; ++mi)
        #pragma unroll
        for (int ni = 0; ni < 4; ++ni)
            #pragma unroll
            for (int j = 0; j < 8; ++j) acc[mi][ni][j] = 0.0f;

    issue(0, 0);

    for (int c = 0; c < totalChunks; ++c) {
        const int b = c & 1;
        if (c + 1 < totalChunks) {
            issue(c + 1, b ^ 1);
            asm volatile("s_wait_asynccnt 4" ::: "memory");  // chunk c complete
        } else {
            asm volatile("s_wait_asynccnt 0" ::: "memory");
        }
        __syncthreads();                                     // cross-wave visibility

        const unsigned short* A_cur = &sA[b][0];
        const unsigned short* B_cur = &sB[b][0];

        // Fragments per ISA 16-bit layout: lane half `khalf` carries k-groups
        // {8*khalf..8*khalf+7} and {16+8*khalf..16+8*khalf+7} -> 2 ds_load_b128.
        FragBF16 a[2];
        #pragma unroll
        for (int mi = 0; mi < 2; ++mi) {
            int m = wm * 32 + mi * 16 + mrow;
            a[mi].q[0] = *(const v4u*)&A_cur[m * BK + 8 * khalf];
            a[mi].q[1] = *(const v4u*)&A_cur[m * BK + 16 + 8 * khalf];
        }
        FragBF16 bfr[4];
        #pragma unroll
        for (int ni = 0; ni < 4; ++ni) {
            int n = wn * 64 + ni * 16 + mrow;
            bfr[ni].q[0] = *(const v4u*)&B_cur[n * BK + 8 * khalf];
            bfr[ni].q[1] = *(const v4u*)&B_cur[n * BK + 16 + 8 * khalf];
        }

        #pragma unroll
        for (int mi = 0; mi < 2; ++mi)
            #pragma unroll
            for (int ni = 0; ni < 4; ++ni)
                acc[mi][ni] = __builtin_amdgcn_wmma_f32_16x16x32_bf16(
                    false, a[mi].v, false, bfr[ni].v,
                    (short)0, acc[mi][ni], false, false);

        __syncthreads();                                     // protect buf reuse
    }

    // Epilogue: C layout -> VGPR v holds M = v + 8*khalf, N = lane%16.
    #pragma unroll
    for (int mi = 0; mi < 2; ++mi) {
        #pragma unroll
        for (int ni = 0; ni < 4; ++ni) {
            #pragma unroll
            for (int v = 0; v < 8; ++v) {
                int m = m0 + wm * 32 + mi * 16 + v + 8 * khalf;
                int n = n0 + wn * 64 + ni * 16 + mrow;
                if (m < N && n < Fout) {
                    float r = acc[mi][ni][v] + bias[n];
                    out[(size_t)m * Fout + n] = r > 0.0f ? r : 0.0f;
                }
            }
        }
    }
}

// ---------------------------------------------------------------------------
// Host-side orchestration
// ---------------------------------------------------------------------------

static inline void* ws_carve(char* base, size_t& off, size_t bytes) {
    off = (off + 255) & ~(size_t)255;
    void* p = base + off;
    off += bytes;
    return p;
}

static inline int round_up(int x, int m) { return ((x + m - 1) / m) * m; }

static inline void launch_fill(float* p, size_t n, hipStream_t s) {
    fill_zero_kernel<<<2048, 256, 0, s>>>(p, n);
}

static void run_layer(const float* in, int Fin, int Fout,
                      const float* W, const float* b, float* out,
                      float* t1, float* t2, unsigned short* Abf, unsigned short* Bbf,
                      const int* src, const int* dst, const float* ew,
                      int N, int Mpad, int E, hipStream_t stream) {
    const int Kp      = round_up(Fin, BK);
    const int FoutPad = round_up(Fout, BN);
    const int spmm_threads = (Fin <= 128) ? 128 : 256;

    // t1 = prop(in)
    launch_fill(t1, (size_t)N * Fin, stream);
    spmm_scatter_kernel<<<E, spmm_threads, 0, stream>>>(in, t1, src, dst, ew, Fin);
    // t2raw = prop(t1)   (cheb tail fused into pack_A)
    launch_fill(t2, (size_t)N * Fin, stream);
    spmm_scatter_kernel<<<E, spmm_threads, 0, stream>>>(t1, t2, src, dst, ew, Fin);

    // Pack operands to zero-padded bf16
    pack_A_kernel<<<4096, 256, 0, stream>>>(in, t1, t2, Abf, N, Mpad, Fin, Kp);
    pack_B_kernel<<<1024, 256, 0, stream>>>(W, Bbf, Fin, Kp, Fout, FoutPad);

    // out = relu([t0|t1|t2] @ [W0;W1;W2] + b)
    dim3 grid(FoutPad / BN, Mpad / BM);
    wmma_gemm_bf16_kernel<<<grid, 256, 0, stream>>>(Abf, Bbf, b, out,
                                                    N, 3 * Kp, Kp, Fout, FoutPad);
}

extern "C" void kernel_launch(void* const* d_in, const int* in_sizes, int n_in,
                              void* d_out, int out_size, void* d_ws, size_t ws_size,
                              hipStream_t stream) {
    const float* x  = (const float*)d_in[0];       // [N,128]
    const int*   ei = (const int*)d_in[1];         // [2,E]
    const float* W1 = (const float*)d_in[2];       // [3,128,250]
    const float* b1 = (const float*)d_in[3];
    const float* W2 = (const float*)d_in[4];       // [3,250,500]
    const float* b2 = (const float*)d_in[5];
    const float* W3 = (const float*)d_in[6];       // [3,500,1000]
    const float* b3 = (const float*)d_in[7];
    float* out = (float*)d_out;

    const int F1 = 128;
    const int N  = in_sizes[0] / F1;
    const int E  = in_sizes[1] / 2;
    const int F2 = in_sizes[3];                    // 250
    const int F3 = in_sizes[5];                    // 500
    const int F4 = in_sizes[7];                    // 1000
    (void)n_in; (void)out_size; (void)ws_size;

    const int Mpad   = round_up(N, BM);
    const int KpMax  = round_up(F3, BK);           // 512 (largest Fin)
    const int FoPMax = round_up(F4, BN);           // 1024 (largest Fout)

    const int* src = ei;
    const int* dst = ei + E;

    char* ws = (char*)d_ws;
    size_t off = 0;
    float* deg = (float*)ws_carve(ws, off, (size_t)N * 4);
    float* dis = (float*)ws_carve(ws, off, (size_t)N * 4);
    float* ew  = (float*)ws_carve(ws, off, (size_t)E * 4);
    float* t1  = (float*)ws_carve(ws, off, (size_t)N * F3 * 4);
    float* t2  = (float*)ws_carve(ws, off, (size_t)N * F3 * 4);
    float* h1  = (float*)ws_carve(ws, off, (size_t)N * F2 * 4);        // layer-1 out
    float* h2  = (float*)ws_carve(ws, off, (size_t)N * F3 * 4);        // layer-2 out
    unsigned short* Abf = (unsigned short*)ws_carve(ws, off, (size_t)Mpad * 3 * KpMax * 2);
    unsigned short* Bbf = (unsigned short*)ws_carve(ws, off, (size_t)3 * FoPMax * KpMax * 2);

    // --- graph normalization: deg -> dis -> edge weights ---
    launch_fill(deg, (size_t)N, stream);
    degree_kernel<<<(E + 255) / 256, 256, 0, stream>>>(src, deg, E);
    dis_kernel<<<(N + 255) / 256, 256, 0, stream>>>(deg, dis, N);
    edge_weight_kernel<<<(E + 255) / 256, 256, 0, stream>>>(src, dst, dis, ew, E);

    // --- three Chebyshev layers ---
    run_layer(x,  F1, F2, W1, b1, h1,  t1, t2, Abf, Bbf, src, dst, ew, N, Mpad, E, stream);
    run_layer(h1, F2, F3, W2, b2, h2,  t1, t2, Abf, Bbf, src, dst, ew, N, Mpad, E, stream);
    run_layer(h2, F3, F4, W3, b3, out, t1, t2, Abf, Bbf, src, dst, ew, N, Mpad, E, stream);
}
